// SymmetricContraction_23373212025193
// MI455X (gfx1250) — matrix operand order repaired and checked by reading, not compiled, not run on hardware
//
#include <hip/hip_runtime.h>

// ---------------------------------------------------------------------------
// MACE symmetric contraction (correlation 3) for MI455X / gfx1250.
//
// out[n,L,c] = sum_k (A3[e,c]@xx + A1bias)[L,k] * x[n,k,c] + (A2[e,c]@xx)[L]
// where A3 = U3*W3 ([256 x 256] per e,c), A2 = U2*W2 (16 extra GEMM rows),
// A1 = U1*W1 folded in as accumulator bias. Heavy part = bf16 WMMA GEMM
// [272 x 256] x [256 x 16-atom-tile] with f32 accumulation.
// ---------------------------------------------------------------------------

typedef __attribute__((ext_vector_type(16))) __bf16 v16bf;
typedef __attribute__((ext_vector_type(8)))  float  v8f;

#define NATOMS  2048
#define LDIM    16
#define C_CH    64
#define E_TYPES 4
#define NLOUT   16
#define NK3     23
#define NK2     5
#define NK1     2
#define MT      17   // 16 order-3 row-tiles (one per L) + 1 order-2 row-tile
#define KC      8    // K = 256 -> 8 chunks of 32 (bf16 WMMA K)

// workspace layout (bytes)
#define WS_CNT   0
#define WS_IDX   256                               // 4*2048*4 = 32768
#define WS_BIAS  33024                             // 4*64*17*256*4 = 4456448
#define WS_ACOMB 4489472                           // 4*64*17*8*512*2 = 35651584

__global__ void zero_counts_k(int* cnt) {
    if (threadIdx.x < E_TYPES) cnt[threadIdx.x] = 0;
}

__global__ void bin_atoms_k(const int* __restrict__ types, int* cnt,
                            int* __restrict__ idxl) {
    int n = blockIdx.x * blockDim.x + threadIdx.x;
    if (n < NATOMS) {
        int t = types[n];
        int s = atomicAdd(&cnt[t], 1);
        idxl[t * NATOMS + s] = n;
    }
}

// Build combined A table in WMMA A-fragment order.
// A-fragment (16x32 bf16, ISA 7.12.2): lane<16 row=lane, elems 0..7->K0..7,
// 8..15->K16..23 ; lane>=16 row=lane-16, elems 0..7->K8..15, 8..15->K24..31.
// One thread per (e, m, kc, lane, elem); inner loop covers all 64 channels.
__global__ void build_A_k(const float* __restrict__ U3, const float* __restrict__ U2,
                          const float* __restrict__ W3, const float* __restrict__ W2,
                          __bf16* __restrict__ Acomb) {
    int tid = blockIdx.x * blockDim.x + threadIdx.x;   // 278528 exact
    int ei = tid & 15;
    int l  = (tid >> 4) & 31;
    int kc = (tid >> 9) & 7;
    int m  = (tid >> 12) % MT;
    int e  = tid / (MT << 12);
    int hi = l >> 4, lr = l & 15;
    int kl = hi ? (ei < 8 ? 8 + ei : 16 + ei)
                : (ei < 8 ? ei     : 8 + ei);
    int kk = kc * 32 + kl;
    int i = kk >> 4, j = kk & 15;

    float acc[64];
#pragma unroll
    for (int c = 0; c < 64; ++c) acc[c] = 0.f;

    if (m < 16) {                 // order-3 rows: row-in-tile = k, L = m
        int off = (((m * 16 + i) * 16 + j) * 16 + lr) * NK3;
        for (int w = 0; w < NK3; ++w) {
            float u = U3[off + w];
            if (u != 0.f) {
                const float* wp = W3 + (e * NK3 + w) * 64;
#pragma unroll
                for (int c = 0; c < 64; ++c) acc[c] += u * wp[c];
            }
        }
    } else {                      // order-2 rows: row-in-tile = L'
        int off = ((lr * 16 + i) * 16 + j) * NK2;
        for (int w = 0; w < NK2; ++w) {
            float u = U2[off + w];
            if (u != 0.f) {
                const float* wp = W2 + (e * NK2 + w) * 64;
#pragma unroll
                for (int c = 0; c < 64; ++c) acc[c] += u * wp[c];
            }
        }
    }
    size_t frag    = (size_t)(m * KC + kc) * 512 + l * 16 + ei;
    size_t cstride = (size_t)MT * KC * 512;            // 69632
    __bf16* base   = Acomb + (size_t)e * 64 * cstride + frag;
#pragma unroll
    for (int c = 0; c < 64; ++c) base[(size_t)c * cstride] = (__bf16)acc[c];
}

// Order-1 bias in C/D fragment order: VGPR r, lane l -> k = r + 8*(l>=16).
__global__ void build_bias_k(const float* __restrict__ U1,
                             const float* __restrict__ W1,
                             float* __restrict__ Bias) {
    int tid = blockIdx.x * blockDim.x + threadIdx.x;   // 1114112 exact
    int l  = tid & 31;
    int r  = (tid >> 5) & 7;
    int m  = (tid >> 8) % MT;
    int ec = tid / (MT * 256);
    int c  = ec & 63, e = ec >> 6;
    float v = 0.f;
    if (m < 16) {
        int k   = r + 8 * (l >> 4);
        int off = (m * 16 + k) * NK1;
        v = U1[off + 0] * W1[(e * NK1 + 0) * 64 + c]
          + U1[off + 1] * W1[(e * NK1 + 1) * 64 + c];
    }
    Bias[tid] = v;
}

// Main GEMM: one wave per (type-tile-slot, channel).
__global__ __launch_bounds__(32) void sym_contract_main_k(
    const float* __restrict__ x, const int* __restrict__ cnt,
    const int* __restrict__ idxl, const __bf16* __restrict__ Acomb,
    const float* __restrict__ Bias, float* __restrict__ out) {
    __shared__ float xlds[256];
    const int l    = threadIdx.x;
    const int slot = blockIdx.x;         // 0..511 (128 tile slots per type)
    const int c    = blockIdx.y;         // 0..63
    const int e    = slot >> 7;
    const int tile = slot & 127;
    const int base = tile * 16;
    const int ce   = cnt[e];
    if (base >= ce) return;              // wave-uniform early exit

    const int lr = l & 15;
    const int hi = l >> 4;

    int myidx = -1;
    if (l < 16 && base + l < ce) myidx = idxl[e * NATOMS + base + l];
    const int colidx = __shfl(myidx, lr, 32);   // atom for this lane's column

    // fill LDS x-tile [16 atoms x 16 i] for this channel
    {
        int n    = l >> 1;
        int nIdx = __shfl(myidx, n, 32);
        int i0   = (l & 1) * 8;
#pragma unroll
        for (int ii = 0; ii < 8; ++ii) {
            float v = 0.f;
            if (nIdx >= 0) v = x[(size_t)nIdx * (LDIM * C_CH) + (i0 + ii) * C_CH + c];
            xlds[n * 16 + i0 + ii] = v;
        }
    }
    __syncthreads();

    float xr[16];
#pragma unroll
    for (int i = 0; i < 16; ++i) xr[i] = xlds[lr * 16 + i];

    // B fragments (32x16 bf16, same K order as A): lane<16 elems 0..7->K0..7,
    // 8..15->K16..23 ; lane>=16 elems 0..7->K8..15, 8..15->K24..31 ; column n = lane & 15.
    v16bf bfrag[KC];
#pragma unroll
    for (int kcI = 0; kcI < KC; ++kcI) {
#pragma unroll
        for (int eidx = 0; eidx < 16; ++eidx) {
            int kk = kcI * 32 + hi * 8 + eidx + (eidx >= 8 ? 8 : 0);
            int i = kk >> 4, j = kk & 15;
            bfrag[kcI][eidx] = (__bf16)(xr[i] * xr[j]);
        }
    }

    const size_t cstride = (size_t)MT * KC * 512;
    const __bf16* aB   = Acomb + (size_t)(e * 64 + c) * cstride;
    const float*  bsB  = Bias  + (size_t)(e * 64 + c) * (MT * 256);

    float sum3[16];
#pragma unroll
    for (int m = 0; m < MT; ++m) {
        v8f acc;
#pragma unroll
        for (int r = 0; r < 8; ++r) acc[r] = bsB[m * 256 + r * 32 + l];
#pragma unroll
        for (int kcI = 0; kcI < KC; ++kcI) {
            v16bf a = *(const v16bf*)(aB + (size_t)(m * KC + kcI) * 512 + l * 16);
            acc = __builtin_amdgcn_wmma_f32_16x16x32_bf16(
                      false, a, false, bfrag[kcI], (short)0, acc, false, false);
        }
        if (m < 16) {
            // out3[L=m, n] partial = sum_k (t3+A1)[L,k]*x[n,k]; halves hold
            // k 0..7 / 8..15 -> reduce across half-waves (replicated result).
            float p = 0.f;
#pragma unroll
            for (int r = 0; r < 8; ++r) {
                float xk = hi ? xr[r + 8] : xr[r];
                p += acc[r] * xk;
            }
            p += __shfl_xor(p, 16, 32);
            sum3[m] = p;
        } else {
            // order-2 tile: element (r,l) = t2'[n=l&15, L=r+8*hi]
#pragma unroll
            for (int r = 0; r < 8; ++r) {
                float tot = acc[r] + (hi ? sum3[r + 8] : sum3[r]);
                if (colidx >= 0) {
                    int L = r + 8 * hi;
                    out[(size_t)colidx * (NLOUT * C_CH) + L * C_CH + c] = tot;
                }
            }
        }
    }
}

extern "C" void kernel_launch(void* const* d_in, const int* in_sizes, int n_in,
                              void* d_out, int out_size, void* d_ws, size_t ws_size,
                              hipStream_t stream) {
    const float* x     = (const float*)d_in[0];
    const int*   types = (const int*)  d_in[1];
    const float* U3    = (const float*)d_in[2];
    const float* U2    = (const float*)d_in[3];
    const float* U1    = (const float*)d_in[4];
    const float* W3    = (const float*)d_in[5];
    const float* W2    = (const float*)d_in[6];
    const float* W1    = (const float*)d_in[7];
    float* out = (float*)d_out;

    char* ws = (char*)d_ws;
    int*    cnt   = (int*)   (ws + WS_CNT);
    int*    idxl  = (int*)   (ws + WS_IDX);
    float*  Bias  = (float*) (ws + WS_BIAS);
    __bf16* Acomb = (__bf16*)(ws + WS_ACOMB);

    zero_counts_k<<<1, 32, 0, stream>>>(cnt);
    bin_atoms_k<<<(NATOMS + 255) / 256, 256, 0, stream>>>(types, cnt, idxl);
    build_A_k<<<278528 / 256, 256, 0, stream>>>(U3, U2, W3, W2, Acomb);
    build_bias_k<<<1114112 / 256, 256, 0, stream>>>(U1, W1, Bias);

    dim3 grid(E_TYPES * 128, C_CH);   // (512 tile slots, 64 channels)
    sym_contract_main_k<<<grid, 32, 0, stream>>>(x, cnt, idxl, Acomb, Bias, out);
}
